// RIM_1958505087272
// MI455X (gfx1250) — matrix operand order repaired and checked by reading, not compiled, not run on hardware
//
#include <hip/hip_runtime.h>

// ---------------- CDNA5 WMMA plumbing (wave32, 16x16x32 f16 -> f32) ----------------
typedef __attribute__((ext_vector_type(16))) _Float16 v16h;
typedef __attribute__((ext_vector_type(8))) float v8f;

#define NPTS 50000
#define BATCH 8
#define CIN 64
#define LTOK 256
#define TCH 128
#define KNN 32
#define T_OFF 51200000L  // offset of second output 't' inside d_out (floats)

union FragU { v16h v; uint4 q[2]; };
union H8 { _Float16 h[8]; uint4 q; };

// A fragment: 16x32 f16, row-major source [m][k], lda in halves.
// lane = (g<<4)|r : row m=r ; halves 0..15 = A[r][g*16 .. g*16+15]
__device__ __forceinline__ v16h frag_a(const _Float16* A, int lda, int lane) {
  int r = lane & 15, g = lane >> 4;
  const _Float16* p = A + (long)r * lda + g * 16;
  FragU u;
  u.q[0] = *(const uint4*)p;
  u.q[1] = *(const uint4*)(p + 8);
  return u.v;
}
// B fragment: 32x16 f16 from B^T storage [n][k], ldb in halves.
// lane = (g<<4)|n : col n ; halves 0..15 = B^T[n][g*16 .. g*16+15]
__device__ __forceinline__ v16h frag_b(const _Float16* BT, int ldb, int lane) {
  int n = lane & 15, g = lane >> 4;
  const _Float16* p = BT + (long)n * ldb + g * 16;
  FragU u;
  u.q[0] = *(const uint4*)p;
  u.q[1] = *(const uint4*)(p + 8);
  return u.v;
}
__device__ __forceinline__ v8f wmma16(v16h a, v16h b, v8f c) {
  return __builtin_amdgcn_wmma_f32_16x16x32_f16(false, a, false, b, (short)0, c, false, false);
}

// ---------------- f32 -> f16 conversion ----------------
__global__ void cvt16_kernel(const float* __restrict__ a, _Float16* __restrict__ o, int n) {
  int i = blockIdx.x * 256 + threadIdx.x;
  if (i < n) o[i] = (_Float16)a[i];
}

// ---------------- KNN gather + max pool -> pooled^T f16 [b][l][c] ----------------
__global__ __launch_bounds__(64) void knnpool_kernel(const float* __restrict__ feat,
                                                     const int* __restrict__ idx,
                                                     _Float16* __restrict__ pooledT) {
  int b = blockIdx.y, l = blockIdx.x, c = threadIdx.x;
  const int* ip = idx + ((long)b * LTOK + l) * KNN;
  const float* f = feat + ((long)b * CIN + c) * NPTS;
  float m = -3.4e38f;
  for (int k = 0; k < KNN; k++) m = fmaxf(m, f[ip[k]]);
  pooledT[((long)b * LTOK + l) * CIN + c] = (_Float16)m;
}

// ---------------- generic WMMA GEMM: D = relu?(scale*(A@B)) + R ----------------
struct GemmP {
  const _Float16* A; const _Float16* BT;
  const float* R; float* Cf; _Float16* CT; _Float16* Cr;
  long sAb, sAh, sBb, sBh, sRb, sRh, sCfb, sCfh, sCTb, sCTh, sCrb, sCrh;
  int lda, ldb, ldr, ldcf, ldct, ldcr;
  int M, N, K, H;
  float scale; int relu;
};

__global__ __launch_bounds__(256) void gemm16_kernel(GemmP p) {
  int y = blockIdx.y;
  int b = y / p.H, h = y - b * p.H;
  int lane = threadIdx.x & 31, wave = threadIdx.x >> 5;
  int mt = p.M >> 4;
  int tiles = mt * (p.N >> 4);
  int tile = blockIdx.x * 8 + wave;
  if (tile >= tiles) return;
  int tm = tile % mt, tn = tile / mt;
  const _Float16* A = p.A + b * p.sAb + h * p.sAh + (long)(tm * 16) * p.lda;
  const _Float16* BT = p.BT + b * p.sBb + h * p.sBh + (long)(tn * 16) * p.ldb;
  v8f acc = {};
  for (int k0 = 0; k0 < p.K; k0 += 32)
    acc = wmma16(frag_a(A + k0, p.lda, lane), frag_b(BT + k0, p.ldb, lane), acc);
  int g = lane >> 4, c = lane & 15;
  int m0 = tm * 16 + g * 8, n = tn * 16 + c;
  float vals[8];
#pragma unroll
  for (int i = 0; i < 8; i++) {
    float v = acc[i] * p.scale;
    if (p.relu) v = fmaxf(v, 0.f);
    if (p.R) v += p.R[b * p.sRb + h * p.sRh + (long)(m0 + i) * p.ldr + n];
    vals[i] = v;
  }
  if (p.Cf) {
    float* o = p.Cf + b * p.sCfb + h * p.sCfh + n;
#pragma unroll
    for (int i = 0; i < 8; i++) o[(long)(m0 + i) * p.ldcf] = vals[i];
  }
  if (p.CT) {  // transposed f16 out [n][m] : one 16B store per lane
    H8 u;
#pragma unroll
    for (int i = 0; i < 8; i++) u.h[i] = (_Float16)vals[i];
    *(uint4*)(p.CT + b * p.sCTb + h * p.sCTh + (long)n * p.ldct + m0) = u.q;
  }
  if (p.Cr) {  // row-major f16 out [m][n]
    _Float16* o = p.Cr + b * p.sCrb + h * p.sCrh + n;
#pragma unroll
    for (int i = 0; i < 8; i++) o[(long)(m0 + i) * p.ldcr] = (_Float16)vals[i];
  }
}

// ---------------- row softmax (token attention): f32 [bh][256][256] -> f16 ----------------
__global__ __launch_bounds__(256) void softmax_rows_kernel(const float* __restrict__ in,
                                                           _Float16* __restrict__ out) {
  int bh = blockIdx.y;
  int wave = threadIdx.x >> 5, lane = threadIdx.x & 31;
  int row = blockIdx.x * 8 + wave;
  const float* r = in + ((long)bh * 256 + row) * 256;
  float v[8], mx = -3.4e38f;
#pragma unroll
  for (int j = 0; j < 8; j++) { v[j] = r[j * 32 + lane]; mx = fmaxf(mx, v[j]); }
  for (int off = 16; off >= 1; off >>= 1) mx = fmaxf(mx, __shfl_xor(mx, off, 32));
  float s = 0.f;
#pragma unroll
  for (int j = 0; j < 8; j++) { v[j] = __expf(v[j] - mx); s += v[j]; }
  for (int off = 16; off >= 1; off >>= 1) s += __shfl_xor(s, off, 32);
  float inv = 1.f / s;
  _Float16* o = out + ((long)bh * 256 + row) * 256;
#pragma unroll
  for (int j = 0; j < 8; j++) o[j * 32 + lane] = (_Float16)(v[j] * inv);
}

// ---------------- fused projector: x = wfb@feat ; pq = wpq@x ;
//  coef = softmax_l(pk.pq/8) ; out = x + pv@coef   (per 64-point tile) ----------------
#define SM_XT   0        // f16 [64][136]  x^T
#define SM_BUF  17408    // f16 [64][136]  (feat^T stride 72, then pq^T stride 136)
#define SM_X    34816    // f32 [128][64]
#define SM_LG   67584    // f32 [256][64]  logits
#define SM_CF   133120   // f16 [64][264]  coef^T (unnormalized exp)
#define SM_RED  166912   // f32 [256]
#define SM_MX   167936   // f32 [64] (max, then 1/sum)
#define SM_TOT  168192

__global__ __launch_bounds__(512) void projector_kernel(
    const float* __restrict__ feat_all, const _Float16* __restrict__ wfb,
    const _Float16* __restrict__ wpq, const _Float16* __restrict__ pkT,
    const _Float16* __restrict__ pv, float* __restrict__ out) {
  extern __shared__ char smem[];
  _Float16* sXT = (_Float16*)(smem + SM_XT);
  _Float16* sBUF = (_Float16*)(smem + SM_BUF);
  float* sX = (float*)(smem + SM_X);
  float* sLG = (float*)(smem + SM_LG);
  _Float16* sCF = (_Float16*)(smem + SM_CF);
  float* sRED = (float*)(smem + SM_RED);
  float* sMX = (float*)(smem + SM_MX);

  int b = blockIdx.y;
  int n0 = blockIdx.x * 64;
  int tid = threadIdx.x, lane = tid & 31, wave = tid >> 5;
  const float* feat = feat_all + (long)b * CIN * NPTS;

  // stage in_feature tile transposed -> sBUF [n][c] (stride 72), f16
  for (int e = tid; e < 64 * 64; e += 512) {
    int cc = e >> 6, n = e & 63;
    int gn = n0 + n; if (gn >= NPTS) gn = NPTS - 1;
    sBUF[n * 72 + cc] = (_Float16)feat[(long)cc * NPTS + gn];
  }
  __syncthreads();

  // x = wfb(128x64) @ X : 32 tiles, K=64
  for (int tile = wave; tile < 32; tile += 16) {
    int tm = tile & 7, tn = tile >> 3;
    v8f acc = {};
    for (int k0 = 0; k0 < 64; k0 += 32)
      acc = wmma16(frag_a(wfb + tm * 16 * 64 + k0, 64, lane),
                   frag_b(sBUF + tn * 16 * 72 + k0, 72, lane), acc);
    int g = lane >> 4, c = lane & 15;
    int m0 = tm * 16 + g * 8, n = tn * 16 + c;
    H8 u;
#pragma unroll
    for (int i = 0; i < 8; i++) { sX[(m0 + i) * 64 + n] = acc[i]; u.h[i] = (_Float16)acc[i]; }
    *(uint4*)(sXT + n * 136 + m0) = u.q;
  }
  __syncthreads();

  // pq = wpq(128x128) @ x : 32 tiles, K=128 -> sBUF as pq^T [n][136]
  for (int tile = wave; tile < 32; tile += 16) {
    int tm = tile & 7, tn = tile >> 3;
    v8f acc = {};
    for (int k0 = 0; k0 < 128; k0 += 32)
      acc = wmma16(frag_a(wpq + tm * 16 * 128 + k0, 128, lane),
                   frag_b(sXT + tn * 16 * 136 + k0, 136, lane), acc);
    int g = lane >> 4, c = lane & 15;
    int m0 = tm * 16 + g * 8, n = tn * 16 + c;
    H8 u;
#pragma unroll
    for (int i = 0; i < 8; i++) u.h[i] = (_Float16)acc[i];
    *(uint4*)(sBUF + n * 136 + m0) = u.q;
  }
  __syncthreads();

  const _Float16* pkTb = pkT + (long)b * LTOK * TCH;
  const _Float16* pvb = pv + (long)b * TCH * LTOK;

  for (int h = 0; h < 2; h++) {
    // logits[l][n] = sum_d pk[d][l]*pq[d][n] : 64 tiles, K=64
    for (int tile = wave; tile < 64; tile += 16) {
      int tm = tile & 15, tn = tile >> 4;
      v8f acc = {};
      for (int k0 = 0; k0 < 64; k0 += 32)
        acc = wmma16(frag_a(pkTb + tm * 16 * TCH + h * 64 + k0, TCH, lane),
                     frag_b(sBUF + tn * 16 * 136 + h * 64 + k0, 136, lane), acc);
      int g = lane >> 4, c = lane & 15;
      int m0 = tm * 16 + g * 8, n = tn * 16 + c;
#pragma unroll
      for (int i = 0; i < 8; i++) sLG[(m0 + i) * 64 + n] = acc[i];
    }
    __syncthreads();

    // softmax over l per column n, scale 1/8; store unnormalized exp in sCF
    int q = tid >> 6, n = tid & 63;
    if (tid < 256) {
      float mx = -3.4e38f;
      for (int j = 0; j < 64; j++) mx = fmaxf(mx, sLG[(q * 64 + j) * 64 + n]);
      sRED[q * 64 + n] = mx;
    }
    __syncthreads();
    if (tid < 64)
      sMX[tid] = fmaxf(fmaxf(sRED[tid], sRED[64 + tid]), fmaxf(sRED[128 + tid], sRED[192 + tid]));
    __syncthreads();
    if (tid < 256) {
      float mx = sMX[n], s = 0.f;
      for (int j = 0; j < 64; j++) {
        int l = q * 64 + j;
        float e = __expf((sLG[l * 64 + n] - mx) * 0.125f);
        sCF[n * 264 + l] = (_Float16)e;
        s += e;
      }
      sRED[q * 64 + n] = s;
    }
    __syncthreads();
    if (tid < 64)
      sMX[tid] = 1.f / (sRED[tid] + sRED[64 + tid] + sRED[128 + tid] + sRED[192 + tid]);
    __syncthreads();

    // proj[d][n] = sum_l pv[h*64+d][l]*exp[l][n]; out = x + proj * inv_sum
    {
      int tm = wave & 3, tn = wave >> 2;  // exactly 16 tiles / 16 waves, K=256
      v8f acc = {};
      for (int k0 = 0; k0 < 256; k0 += 32)
        acc = wmma16(frag_a(pvb + (h * 64 + tm * 16) * LTOK + k0, LTOK, lane),
                     frag_b(sCF + tn * 16 * 264 + k0, 264, lane), acc);
      int g = lane >> 4, c = lane & 15;
      int m0 = tm * 16 + g * 8, nn = tn * 16 + c;
      int gn = n0 + nn;
      if (gn < NPTS) {
#pragma unroll
        for (int i = 0; i < 8; i++) {
          int d = h * 64 + m0 + i;
          out[(long)b * TCH * NPTS + (long)d * NPTS + gn] = sX[d * 64 + nn] + acc[i] * sMX[nn];
        }
      }
    }
    __syncthreads();
  }
}

// ---------------------------------------------------------------------------------
extern "C" void kernel_launch(void* const* d_in, const int* in_sizes, int n_in,
                              void* d_out, int out_size, void* d_ws, size_t ws_size,
                              hipStream_t stream) {
  (void)in_sizes; (void)n_in; (void)out_size;
  const float* in_feature = (const float*)d_in[0];
  const float* in_tokens = (const float*)d_in[1];
  const int* knn_idx = (const int*)d_in[2];
  const float* w_dyn1 = (const float*)d_in[3];
  const float* w_dyn2 = (const float*)d_in[4];
  const float* w_fb = (const float*)d_in[5];
  const float* w_k = (const float*)d_in[6];
  const float* w_q = (const float*)d_in[7];
  const float* w_v = (const float*)d_in[8];
  const float* w_ff1 = (const float*)d_in[9];
  const float* w_ff2 = (const float*)d_in[10];
  const float* w_pv = (const float*)d_in[11];
  const float* w_pk = (const float*)d_in[12];
  const float* w_pq = (const float*)d_in[13];
  float* out = (float*)d_out;

  char* ws = (char*)d_ws;
  size_t off = 0;
  auto alloc = [&](size_t bytes) { size_t o = off; off += (bytes + 255) & ~(size_t)255; return o; };

  // f16 weights
  size_t o_wd1 = alloc(8192 * 2), o_wd2 = alloc(16384 * 2), o_wfb = alloc(8192 * 2);
  size_t o_wk = alloc(8192 * 2), o_wq = alloc(8192 * 2), o_wv = alloc(16384 * 2);
  size_t o_wf1 = alloc(32768 * 2), o_wf2 = alloc(32768 * 2);
  size_t o_wpv = alloc(16384 * 2), o_wpk = alloc(16384 * 2), o_wpq = alloc(16384 * 2);
  // intermediates
  size_t o_poolT = alloc((size_t)BATCH * LTOK * CIN * 2);     // f16 [b][l][64]
  size_t o_hidT = alloc((size_t)BATCH * LTOK * TCH * 2);      // f16 [b][l][128]
  size_t o_tok = alloc((size_t)BATCH * TCH * LTOK * 4);       // f32 [b][128][256]
  size_t o_tokT = alloc((size_t)BATCH * LTOK * TCH * 2);      // f16 [b][l][128]
  size_t o_kT = alloc((size_t)BATCH * LTOK * 64 * 2);         // f16 [b][l][64]
  size_t o_qT = alloc((size_t)BATCH * LTOK * 64 * 2);
  size_t o_v16 = alloc((size_t)BATCH * TCH * LTOK * 2);       // f16 [b][128][256] row-major
  size_t o_lgT = alloc((size_t)BATCH * 2 * 256 * 256 * 4);    // f32 [bh][m][n]
  size_t o_attT = alloc((size_t)BATCH * 2 * 256 * 256 * 2);   // f16 [bh][m][n]
  size_t o_t1 = alloc((size_t)BATCH * TCH * LTOK * 4);        // f32
  size_t o_t1T = alloc((size_t)BATCH * LTOK * TCH * 2);       // f16
  size_t o_h2T = alloc((size_t)BATCH * LTOK * 256 * 2);       // f16 [b][m][256]
  size_t o_tT = alloc((size_t)BATCH * LTOK * TCH * 2);        // f16
  size_t o_pv16 = alloc((size_t)BATCH * TCH * LTOK * 2);      // f16 row-major
  size_t o_pkT = alloc((size_t)BATCH * LTOK * TCH * 2);       // f16 [b][l][128]
  if (off > ws_size) return;  // workspace too small (≈14.5 MB needed)

  auto F16 = [&](size_t o) { return (_Float16*)(ws + o); };

  // 1) convert weights to f16
  struct CV { const float* s; size_t o; int n; } cv[] = {
      {w_dyn1, o_wd1, 8192}, {w_dyn2, o_wd2, 16384}, {w_fb, o_wfb, 8192},
      {w_k, o_wk, 8192}, {w_q, o_wq, 8192}, {w_v, o_wv, 16384},
      {w_ff1, o_wf1, 32768}, {w_ff2, o_wf2, 32768},
      {w_pv, o_wpv, 16384}, {w_pk, o_wpk, 16384}, {w_pq, o_wpq, 16384}};
  for (auto& c : cv)
    cvt16_kernel<<<(c.n + 255) / 256, 256, 0, stream>>>(c.s, F16(c.o), c.n);

  // 2) KNN gather + maxpool -> pooled^T
  knnpool_kernel<<<dim3(LTOK, BATCH), 64, 0, stream>>>(in_feature, knn_idx, F16(o_poolT));

  auto launch = [&](GemmP p) {
    int tiles = (p.M >> 4) * (p.N >> 4);
    gemm16_kernel<<<dim3((tiles + 7) / 8, BATCH * p.H), 256, 0, stream>>>(p);
  };
  auto G = [&](size_t oA, int lda, long sAb, long sAh, size_t oB, int ldb, long sBb, long sBh,
               int M, int N, int K, int H, float scale, int relu) {
    GemmP p{};
    p.A = F16(oA); p.lda = lda; p.sAb = sAb; p.sAh = sAh;
    p.BT = F16(oB); p.ldb = ldb; p.sBb = sBb; p.sBh = sBh;
    p.M = M; p.N = N; p.K = K; p.H = H; p.scale = scale; p.relu = relu;
    return p;
  };

  // 3) hid = relu(w_dyn1 @ pooled)  -> hidT
  { GemmP p = G(o_wd1, 64, 0, 0, o_poolT, 64, LTOK * 64L, 0, 128, 256, 64, 1, 1.f, 1);
    p.CT = F16(o_hidT); p.ldct = 128; p.sCTb = LTOK * 128L; launch(p); }
  // 4) tokens = w_dyn2 @ hid + in_tokens -> tokens f32 + tokensT f16
  { GemmP p = G(o_wd2, 128, 0, 0, o_hidT, 128, LTOK * 128L, 0, 128, 256, 128, 1, 1.f, 0);
    p.R = in_tokens; p.ldr = 256; p.sRb = 128L * 256;
    p.Cf = (float*)(ws + o_tok); p.ldcf = 256; p.sCfb = 128L * 256;
    p.CT = F16(o_tokT); p.ldct = 128; p.sCTb = LTOK * 128L; launch(p); }
  // 5) kT = (w_k @ tokens)^T ; 6) qT
  { GemmP p = G(o_wk, 128, 0, 0, o_tokT, 128, LTOK * 128L, 0, 64, 256, 128, 1, 1.f, 0);
    p.CT = F16(o_kT); p.ldct = 64; p.sCTb = LTOK * 64L; launch(p); }
  { GemmP p = G(o_wq, 128, 0, 0, o_tokT, 128, LTOK * 128L, 0, 64, 256, 128, 1, 1.f, 0);
    p.CT = F16(o_qT); p.ldct = 64; p.sCTb = LTOK * 64L; launch(p); }
  // 7) v (f16 row-major)
  { GemmP p = G(o_wv, 128, 0, 0, o_tokT, 128, LTOK * 128L, 0, 128, 256, 128, 1, 1.f, 0);
    p.Cr = F16(o_v16); p.ldcr = 256; p.sCrb = 128L * 256; launch(p); }
  // 8) logitsT[m][n] = sum_d q[d][m] k[d][n] / 16  (per b,h)
  { GemmP p = G(o_qT, 64, LTOK * 64L, 32, o_kT, 64, LTOK * 64L, 32, 256, 256, 32, 2,
                1.f / 16.f, 0);
    p.Cf = (float*)(ws + o_lgT); p.ldcf = 256; p.sCfb = 2L * 65536; p.sCfh = 65536; launch(p); }
  // 9) attnT = rowwise softmax(logitsT)
  softmax_rows_kernel<<<dim3(32, BATCH * 2), 256, 0, stream>>>((const float*)(ws + o_lgT),
                                                               F16(o_attT));
  // 10) t1 = tokens + v @ attn (per b,h) ; also t1T
  { GemmP p = G(o_v16, 256, 128L * 256, 64L * 256, o_attT, 256, 2L * 65536, 65536,
                64, 256, 256, 2, 1.f, 0);
    p.R = (const float*)(ws + o_tok); p.ldr = 256; p.sRb = 128L * 256; p.sRh = 64L * 256;
    p.Cf = (float*)(ws + o_t1); p.ldcf = 256; p.sCfb = 128L * 256; p.sCfh = 64L * 256;
    p.CT = F16(o_t1T); p.ldct = 128; p.sCTb = LTOK * 128L; p.sCTh = 64; launch(p); }
  // 11) hid2 = relu(w_ff1 @ t1) -> hid2T
  { GemmP p = G(o_wf1, 128, 0, 0, o_t1T, 128, LTOK * 128L, 0, 256, 256, 128, 1, 1.f, 1);
    p.CT = F16(o_h2T); p.ldct = 256; p.sCTb = LTOK * 256L; launch(p); }
  // 12) t = w_ff2 @ hid2 + t1 -> d_out tail (f32) + tT
  { GemmP p = G(o_wf2, 256, 0, 0, o_h2T, 256, LTOK * 256L, 0, 128, 256, 256, 1, 1.f, 0);
    p.R = (const float*)(ws + o_t1); p.ldr = 256; p.sRb = 128L * 256;
    p.Cf = out + T_OFF; p.ldcf = 256; p.sCfb = 128L * 256;
    p.CT = F16(o_tT); p.ldct = 128; p.sCTb = LTOK * 128L; launch(p); }
  // 13) pv = w_pv @ t (f16 row-major) ; 14) pkT = (w_pk @ t)^T
  { GemmP p = G(o_wpv, 128, 0, 0, o_tT, 128, LTOK * 128L, 0, 128, 256, 128, 1, 1.f, 0);
    p.Cr = F16(o_pv16); p.ldcr = 256; p.sCrb = 128L * 256; launch(p); }
  { GemmP p = G(o_wpk, 128, 0, 0, o_tT, 128, LTOK * 128L, 0, 128, 256, 128, 1, 1.f, 0);
    p.CT = F16(o_pkT); p.ldct = 128; p.sCTb = LTOK * 128L; launch(p); }

  // 15) fused projector over all N points
  hipFuncSetAttribute((const void*)projector_kernel,
                      hipFuncAttributeMaxDynamicSharedMemorySize, SM_TOT);
  projector_kernel<<<dim3((NPTS + 63) / 64, BATCH), 512, SM_TOT, stream>>>(
      in_feature, F16(o_wfb), F16(o_wpq), F16(o_pkT), F16(o_pv16), out);
}
